// VectorQuantizerEMA_26938034881320
// MI455X (gfx1250) — matrix operand order repaired and checked by reading, not compiled, not run on hardware
//
#include <hip/hip_runtime.h>

#define K_CODES 1024
#define DIM 128
#define NVEC 65536
#define DECAYF 0.99f
#define EPSV 1e-5f
#define THRESH 2.0f
#define CCOST 0.25f

typedef __attribute__((ext_vector_type(16))) __bf16 v16bf;
typedef __attribute__((ext_vector_type(8)))  float  v8f;

// ---------------------------------------------------------------- prep
__global__ void prep_kernel(const float* __restrict__ emb, float* enorm,
                            float* counts, float* embed_sum,
                            float* sse, float* nsum) {
    int i = blockIdx.x * blockDim.x + threadIdx.x;
    if (i < K_CODES * DIM) embed_sum[i] = 0.0f;
    if (i < K_CODES) {
        const float* e = emb + i * DIM;
        float s = 0.0f;
        for (int d = 0; d < DIM; ++d) { float v = e[d]; s += v * v; }
        enorm[i]  = s;
        counts[i] = 0.0f;
    }
    if (i == 0) { *sse = 0.0f; *nsum = 0.0f; }
}

// ------------------------------------------------- pack B = E^T fragments
// B fragment as packed here (contiguous k, same order as A): lane n = l%16 holds code column n,
// lanes 0-15 hold K=kc*32+0..15, lanes 16-31 hold K=kc*32+16..31.
__global__ __launch_bounds__(32) void bfrag_kernel(const float* __restrict__ emb,
                                                   v16bf* __restrict__ bfrag) {
    const int nc   = blockIdx.x >> 2;
    const int kc   = blockIdx.x & 3;
    const int lane = threadIdx.x;
    const int n    = nc * 16 + (lane & 15);
    const int kb   = kc * 32 + ((lane >> 4) & 1) * 16;
    const float* src = emb + n * DIM + kb;
    v16bf v;
#pragma unroll
    for (int j = 0; j < 16; ++j) v[j] = (__bf16)src[j];
    bfrag[blockIdx.x * 32 + lane] = v;
}

// ---------------------------------------------------- fused GEMM + argmin
// 8 waves/block, each owns a 16-row tile of z. B fragments are staged into
// LDS once per block via GLOBAL_LOAD_ASYNC_TO_LDS_B128 (double buffered,
// ASYNCcnt-tracked), then consumed by all waves with ds_load_b128.
#define STAGE_BYTES 32768   // 8 nc-chunks * 4 kc * 32 lanes * 32 B

__global__ __launch_bounds__(256) void argmin_kernel(const float* __restrict__ z,
                                                     const float* __restrict__ enorm,
                                                     const char* __restrict__ bfrag_g,
                                                     int* __restrict__ idx_ws) {
    __shared__ char s_b[2 * STAGE_BYTES];

    const int lane    = threadIdx.x & 31;
    const int wave    = threadIdx.x >> 5;
    const int rowbase = blockIdx.x * 128 + wave * 16;
    const int r       = rowbase + (lane & 15);
    const int khalf   = ((lane >> 4) & 1) * 16;

    // A fragment as loaded here (contiguous k, same order as B): lane holds row m=l%16;
    // lanes 0-15: K = 0..15; lanes 16-31: K = 16..31.
    v16bf a[4];
    const float* zr = z + (size_t)r * DIM;
#pragma unroll
    for (int kc = 0; kc < 4; ++kc) {
        const int kb = kc * 32 + khalf;
#pragma unroll
        for (int j = 0; j < 8; ++j) a[kc][j]     = (__bf16)zr[kb + j];
#pragma unroll
        for (int j = 0; j < 8; ++j) a[kc][8 + j] = (__bf16)zr[kb + 8 + j];
    }

    float best[8]; int bidx[8];
#pragma unroll
    for (int j = 0; j < 8; ++j) { best[j] = 3.4e38f; bidx[j] = 0; }

    const uint32_t lds_base = (uint32_t)(uintptr_t)(&s_b[0]);
    const uint64_t gbase    = (uint64_t)(uintptr_t)bfrag_g;

    // stage chunk 0 into buffer 0
#pragma unroll
    for (int t = 0; t < 8; ++t) {
        const uint32_t boff   = (uint32_t)(t * 256 + threadIdx.x) * 16u;
        const uint32_t ldsoff = lds_base + boff;
        const uint64_t gaddr  = gbase + boff;
        asm volatile("global_load_async_to_lds_b128 %0, %1, off"
                     :: "v"(ldsoff), "v"(gaddr) : "memory");
    }
    asm volatile("s_wait_asynccnt 0x0" ::: "memory");
    __syncthreads();

    for (int outer = 0; outer < 8; ++outer) {
        const uint32_t cur = (uint32_t)(outer & 1) * STAGE_BYTES;
        // prefetch next chunk into the other buffer (async, no wait yet)
        if (outer + 1 < 8) {
            const uint32_t nxt   = (uint32_t)((outer + 1) & 1) * STAGE_BYTES;
            const uint64_t gnext = gbase + (uint32_t)(outer + 1) * (uint32_t)STAGE_BYTES;
#pragma unroll
            for (int t = 0; t < 8; ++t) {
                const uint32_t boff   = (uint32_t)(t * 256 + threadIdx.x) * 16u;
                const uint32_t ldsoff = lds_base + nxt + boff;
                const uint64_t gaddr  = gnext + boff;
                asm volatile("global_load_async_to_lds_b128 %0, %1, off"
                             :: "v"(ldsoff), "v"(gaddr) : "memory");
            }
        }

        // consume current buffer: 8 code-chunks of 16 codes each
        for (int nci = 0; nci < 8; ++nci) {
            v8f c = {};
#pragma unroll
            for (int kc = 0; kc < 4; ++kc) {
                const v16bf b = *(const v16bf*)(&s_b[cur + (size_t)((nci * 4 + kc) * 32 + lane) * 32]);
                c = __builtin_amdgcn_wmma_f32_16x16x32_bf16(
                        false, a[kc], false, b, (short)0, c, false, false);
            }
            const int   n  = (outer * 8 + nci) * 16 + (lane & 15);
            const float en = enorm[n];
#pragma unroll
            for (int j = 0; j < 8; ++j) {
                float cand = en - 2.0f * c[j];
                bool take  = cand < best[j];
                best[j] = take ? cand : best[j];
                bidx[j] = take ? n    : bidx[j];
            }
        }

        // next buffer ready + everyone done reading current before overwrite
        asm volatile("s_wait_asynccnt 0x0" ::: "memory");
        __syncthreads();
    }

    // reduce across the 16 lanes that share each output row (wave32 butterfly)
#pragma unroll
    for (int mask = 8; mask >= 1; mask >>= 1) {
#pragma unroll
        for (int j = 0; j < 8; ++j) {
            float ov = __shfl_xor(best[j], mask, 32);
            int   oi = __shfl_xor(bidx[j], mask, 32);
            bool take = (ov < best[j]) || (ov == best[j] && oi < bidx[j]);
            best[j] = take ? ov : best[j];
            bidx[j] = take ? oi : bidx[j];
        }
    }

    if ((lane & 15) == 0) {
        const int moff = (lane >> 4) * 8;   // lane 0 -> rows 0..7, lane 16 -> rows 8..15
#pragma unroll
        for (int j = 0; j < 8; ++j) idx_ws[rowbase + moff + j] = bidx[j];
    }
}

// ------------------------------------ gather z_q, loss SSE, scatter stats
__global__ __launch_bounds__(128) void gather_kernel(const float* __restrict__ z,
                                                     const float* __restrict__ emb,
                                                     const int* __restrict__ idx_ws,
                                                     float* zq_out, float* idxf_out,
                                                     float* counts, float* embed_sum,
                                                     float* sse) {
    __shared__ float red[128];
    const int row = blockIdx.x;
    const int d   = threadIdx.x;
    const int idx = idx_ws[row];
    const float zv = z[(size_t)row * DIM + d];
    const float ev = emb[idx * DIM + d];
    zq_out[(size_t)row * DIM + d] = ev;   // z + sg(z_q - z) == z_q numerically
    const float diff = ev - zv;
    red[d] = diff * diff;
    atomicAdd(&embed_sum[idx * DIM + d], zv);
    __syncthreads();
    for (int s = 64; s > 0; s >>= 1) {
        if (d < s) red[d] += red[d + s];
        __syncthreads();
    }
    if (d == 0) {
        atomicAdd(sse, red[0]);
        atomicAdd(&counts[idx], 1.0f);
        idxf_out[row] = (float)idx;
    }
}

// ------------------------------------------------------------- EMA stage
__global__ void ema1_kernel(const float* __restrict__ cluster,
                            const float* __restrict__ embed_avg,
                            const float* __restrict__ counts,
                            const float* __restrict__ embed_sum,
                            float* nea_out, float* ncs_ws, float* nsum) {
    int i = blockIdx.x * blockDim.x + threadIdx.x;
    if (i >= K_CODES * DIM) return;
    float nea = DECAYF * embed_avg[i] + (1.0f - DECAYF) * embed_sum[i];
    nea_out[i] = nea;
    if ((i & (DIM - 1)) == 0) {
        int k = i / DIM;
        float ncs = DECAYF * cluster[k] + (1.0f - DECAYF) * counts[k];
        ncs_ws[k] = ncs;
        atomicAdd(nsum, ncs);
    }
}

// ------------------------------------------- dead-code rank (prefix scan)
__global__ __launch_bounds__(1024) void rank_kernel(const float* __restrict__ counts,
                                                    int* __restrict__ rank_ws) {
    __shared__ int scan[K_CODES];
    const int k = threadIdx.x;
    scan[k] = (counts[k] < THRESH) ? 1 : 0;
    __syncthreads();
    for (int off = 1; off < K_CODES; off <<= 1) {
        int v = (k >= off) ? scan[k - off] : 0;
        __syncthreads();
        scan[k] += v;
        __syncthreads();
    }
    rank_ws[k] = scan[k] - 1;
}

// ---------------------------------------------------------------- finish
__global__ void finalize_kernel(const float* __restrict__ z,
                                const float* __restrict__ counts,
                                const float* __restrict__ ncs_ws,
                                const int* __restrict__ rank_ws,
                                const float* __restrict__ nsum_p,
                                const float* __restrict__ sse_p,
                                float* emb_out, float* cs_out,
                                float* nea_out, float* loss_out) {
    int i = blockIdx.x * blockDim.x + threadIdx.x;
    if (i >= K_CODES * DIM) return;
    const int k = i / DIM, d = i & (DIM - 1);
    const float cnt  = counts[k];
    const bool  dead = cnt < THRESH;
    const float ncs  = ncs_ws[k];
    const float ns   = *nsum_p;
    const float smoothed = (ncs + EPSV) / (ns + (float)K_CODES * EPSV) * ns;
    float emb;
    if (dead) {
        int r = rank_ws[k];
        r = r < 0 ? 0 : (r > K_CODES - 1 ? K_CODES - 1 : r);
        unsigned pr = ((unsigned)r * 40503u) & (NVEC - 1);  // stand-in permutation
        float repl = z[(size_t)pr * DIM + d];
        emb = repl;
        nea_out[i] = repl;
    } else {
        emb = nea_out[i] / smoothed;
    }
    emb_out[i] = emb;
    if (d == 0) cs_out[k] = dead ? THRESH : ncs;
    if (i == 0) *loss_out = CCOST * (*sse_p) / (float)((size_t)NVEC * DIM);
}

// ---------------------------------------------------------------- launch
extern "C" void kernel_launch(void* const* d_in, const int* in_sizes, int n_in,
                              void* d_out, int out_size, void* d_ws, size_t ws_size,
                              hipStream_t stream) {
    const float* z         = (const float*)d_in[0];
    const float* emb       = (const float*)d_in[1];
    const float* cluster   = (const float*)d_in[2];
    const float* embed_avg = (const float*)d_in[3];

    char* ws = (char*)d_ws;
    float* enorm     = (float*)(ws + 0);          // 4 KB
    v16bf* bfrag     = (v16bf*)(ws + 4096);       // 256 KB
    int*   idx_ws    = (int*)  (ws + 266240);     // 256 KB
    float* counts    = (float*)(ws + 528384);     // 4 KB
    float* embed_sum = (float*)(ws + 532480);     // 512 KB
    float* ncs_ws    = (float*)(ws + 1056768);    // 4 KB
    int*   rank_ws   = (int*)  (ws + 1060864);    // 4 KB
    float* sse       = (float*)(ws + 1064960);
    float* nsum      = (float*)(ws + 1064964);

    float* out      = (float*)d_out;
    float* zq_out   = out;                  // [16,4096,128] = 8388608
    float* idxf_out = out + 8388608;        // [16,4096]     = 65536
    float* loss_out = out + 8454144;        // scalar
    float* emb_out  = out + 8454145;        // [1024,128]    = 131072
    float* cs_out   = out + 8585217;        // [1024]
    float* nea_out  = out + 8586241;        // [1024,128]    = 131072

    prep_kernel    <<<512, 256, 0, stream>>>(emb, enorm, counts, embed_sum, sse, nsum);
    bfrag_kernel   <<<256, 32, 0, stream>>>(emb, bfrag);
    argmin_kernel  <<<512, 256, 0, stream>>>(z, enorm, (const char*)bfrag, idx_ws);
    gather_kernel  <<<65536, 128, 0, stream>>>(z, emb, idx_ws, zq_out, idxf_out,
                                               counts, embed_sum, sse);
    ema1_kernel    <<<512, 256, 0, stream>>>(cluster, embed_avg, counts, embed_sum,
                                             nea_out, ncs_ws, nsum);
    rank_kernel    <<<1, 1024, 0, stream>>>(counts, rank_ws);
    finalize_kernel<<<512, 256, 0, stream>>>(z, counts, ncs_ws, rank_ws, nsum, sse,
                                             emb_out, cs_out, nea_out, loss_out);
}
